// PointNetSetAbstraction_27041114096409
// MI455X (gfx1250) — matrix operand order repaired and checked by reading, compiled, not run on hardware
//
#include <hip/hip_runtime.h>
#include <hip/hip_bf16.h>

// PointNet++ Set Abstraction for MI455X (gfx1250, wave32).
// Phases: FPS -> ball query -> gather + 3x (1x1 conv + BN + ReLU) via WMMA -> masked maxpool.
// MLP GEMMs use v_wmma_f32_16x16x32_f16 (f16 inputs, f32 accumulate).
// Loop nesting keeps B fragments register-resident (nt outer) to avoid scratch spills.

#define BB 4
#define NN 8192
#define SS 2048
#define KK 64
#define RADIUS 0.2f
#define EPSBN 1e-5f

typedef __attribute__((ext_vector_type(16))) _Float16 v16h;
typedef __attribute__((ext_vector_type(8)))  _Float16 v8h;
typedef __attribute__((ext_vector_type(8)))  float    v8f;

union frag_u { v16h v; v8h h[2]; };

__device__ __forceinline__ v8f wmma32(v16h a, v16h b, v8f c) {
    // D = A(16x32 f16) * B(32x16 f16) + C(16x16 f32)
    return __builtin_amdgcn_wmma_f32_16x16x32_f16(false, a, false, b, (short)0, c, false, false);
}

// A-fragment (16x32 f16) from LDS tile, row-major, stride in halves (multiple of 8).
// Lane L: row = base row + (L&15); element i<8 -> K = ks*32 + (L>>4)*8 + i,
//          element i>=8 -> K = ks*32 + 16 + (L>>4)*8 + (i-8).  (ISA 7.12.2 layout)
__device__ __forceinline__ v16h load_afrag(const _Float16* src, int row, int stride, int ks, int hs) {
    frag_u u;
    const _Float16* p = src + row * stride + ks * 32;
    u.h[0] = *reinterpret_cast<const v8h*>(p + hs * 8);
    u.h[1] = *reinterpret_cast<const v8h*>(p + 16 + hs * 8);
    return u.v;
}

// B-fragment (32x16 f16) pre-packed in global memory: lane L's 16 halves contiguous.
__device__ __forceinline__ v16h load_bfrag(const _Float16* wf, int tile, int lane) {
    frag_u u;
    const _Float16* p = wf + ((size_t)tile * 32 + lane) * 16;
    u.h[0] = *reinterpret_cast<const v8h*>(p);
    u.h[1] = *reinterpret_cast<const v8h*>(p + 8);
    return u.v;
}

// ---------------------------------------------------------------------------
// Kernel 0: pack Conv weights (f32 [Cout,Cin]) into f16 WMMA B-fragment order.
// Tiles: layer1 nt(4)*ks(3)=12, layer2 nt(4)*ks(2)=8 (off 12), layer3 nt(8)*ks(2)=16 (off 20).
// B layout: lane L holds column n = L&15; element i -> K = ks*32 + (i>>3)*16 + (L>>4)*8 + (i&7).
// ---------------------------------------------------------------------------
__global__ void pack_weights_kernel(const float* __restrict__ W1,
                                    const float* __restrict__ W2,
                                    const float* __restrict__ W3,
                                    _Float16* __restrict__ wf) {
    int t = blockIdx.x * 256 + threadIdx.x;           // one thread per packed half
    if (t >= 36 * 512) return;
    int tile = t >> 9;                                 // 0..35
    int lane = (t >> 4) & 31;
    int i    = t & 15;
    int nt, ks, cin; const float* W;
    if (tile < 12)      { nt = tile / 3;        ks = tile % 3;        cin = 67; W = W1; }
    else if (tile < 20) { nt = (tile - 12) / 2; ks = (tile - 12) % 2; cin = 64; W = W2; }
    else                { nt = (tile - 20) / 2; ks = (tile - 20) % 2; cin = 64; W = W3; }
    int n = nt * 16 + (lane & 15);
    int k = ks * 32 + (i >> 3) * 16 + (lane >> 4) * 8 + (i & 7);
    float val = (k < cin) ? W[n * cin + k] : 0.0f;
    wf[t] = (_Float16)val;
}

// ---------------------------------------------------------------------------
// Kernel 1: farthest point sampling. One 1024-thread block per batch.
// dist[] lives in LDS; two-level argmax (shuffle within wave32, LDS across 32 waves).
// ---------------------------------------------------------------------------
__global__ void fps_kernel(const float* __restrict__ x, int* __restrict__ idx_fps) {
    extern __shared__ char smem[];
    float* dist = (float*)smem;                        // NN floats
    float* wval = (float*)(smem + NN * 4);             // 32
    int*   widx = (int*)  (smem + NN * 4 + 32 * 4);    // 32
    int*   sFar = (int*)  (smem + NN * 4 + 32 * 8);    // 1

    const int b   = blockIdx.x;
    const int tid = threadIdx.x;
    const float* xb = x + (size_t)b * NN * 3;

    for (int p = tid; p < NN; p += 1024) dist[p] = 1e10f;
    if (tid == 0) sFar[0] = 0;
    __syncthreads();

    for (int s = 0; s < SS; ++s) {
        int far = sFar[0];
        if (tid == 0) idx_fps[b * SS + s] = far;       // emit BEFORE update (matches scan)
        float cx = xb[far * 3 + 0];
        float cy = xb[far * 3 + 1];
        float cz = xb[far * 3 + 2];

        float bv = -1.0f; int bi = 0;
        for (int p = tid; p < NN; p += 1024) {
            float dx = xb[p * 3 + 0] - cx;
            float dy = xb[p * 3 + 1] - cy;
            float dz = xb[p * 3 + 2] - cz;
            float d  = dx * dx + dy * dy + dz * dz;
            float nd = fminf(dist[p], d);
            dist[p] = nd;
            if (nd > bv) { bv = nd; bi = p; }          // p ascending -> first max kept
        }
        #pragma unroll
        for (int off = 16; off > 0; off >>= 1) {
            float ov = __shfl_down(bv, off, 32);
            int   oi = __shfl_down(bi, off, 32);
            if (ov > bv || (ov == bv && oi < bi)) { bv = ov; bi = oi; }
        }
        int wid = tid >> 5;
        if ((tid & 31) == 0) { wval[wid] = bv; widx[wid] = bi; }
        __syncthreads();
        if (wid == 0) {
            bv = wval[tid & 31]; bi = widx[tid & 31];
            #pragma unroll
            for (int off = 16; off > 0; off >>= 1) {
                float ov = __shfl_down(bv, off, 32);
                int   oi = __shfl_down(bi, off, 32);
                if (ov > bv || (ov == bv && oi < bi)) { bv = ov; bi = oi; }
            }
            if (tid == 0) sFar[0] = bi;
        }
        __syncthreads();
    }
}

// ---------------------------------------------------------------------------
// Kernel 2: ball query (first K indices in ascending order within radius).
// One wave32 per centroid; ballot + prefix-popcount order-preserving compaction.
// Also writes the centroid coordinates output.
// ---------------------------------------------------------------------------
__global__ void ballq_kernel(const float* __restrict__ x,
                             const int* __restrict__ idx_fps,
                             int* __restrict__ idx_ball,
                             float* __restrict__ out_cen) {
    const int lane = threadIdx.x & 31;
    const int cidx = blockIdx.x * 8 + (threadIdx.x >> 5);   // 8 waves per block
    const int b = cidx / SS;
    const float* xb = x + (size_t)b * NN * 3;

    int fp = idx_fps[cidx];
    float cx = xb[fp * 3 + 0];
    float cy = xb[fp * 3 + 1];
    float cz = xb[fp * 3 + 2];
    if (lane < 3) out_cen[(size_t)cidx * 3 + lane] = xb[fp * 3 + lane];

    int* ob = idx_ball + (size_t)cidx * KK;
    int total = 0;
    const float r2 = RADIUS * RADIUS;
    for (int c = 0; c < NN; c += 32) {
        int j = c + lane;
        float dx = xb[j * 3 + 0] - cx;
        float dy = xb[j * 3 + 1] - cy;
        float dz = xb[j * 3 + 2] - cz;
        bool pred = (dx * dx + dy * dy + dz * dz) < r2;
        unsigned msk = (unsigned)__ballot(pred);
        int pos = total + __popc(msk & ((1u << lane) - 1u));
        if (pred && pos < KK) ob[pos] = j;
        total += __popc(msk);
        if (total >= KK) break;                         // uniform across wave
    }
    for (int p = min(total, KK) + lane; p < KK; p += 32) ob[p] = -1;
}

// ---------------------------------------------------------------------------
// Kernel 3: gather + 3-layer MLP (WMMA) + masked max-pool.
// 2 waves per block, one neighborhood (b,s) per wave; 32000 B LDS per wave:
//   Fa 64x104 f16 (13312B) | H1 64x72 f16 (9216B) | H2 64x72 f16 (9216B) | rowmask 64 f32 (256B)
// GEMM nesting: nt (output-channel tile) OUTER and rolled -> only 2-3 B fragments
// live in VGPRs, reused across the 4 unrolled mt (neighbor-row) tiles. A fragments
// stream from LDS (ds_load_b128 x2 each). This avoids the scratch spills seen when
// all B fragments were hoisted.
// ---------------------------------------------------------------------------
__global__ void mlp_kernel(const float* __restrict__ x,
                           const float* __restrict__ xc,
                           const int* __restrict__ idx_fps,
                           const int* __restrict__ idx_ball,
                           const _Float16* __restrict__ wf,
                           const float* __restrict__ b1, const float* __restrict__ g1,
                           const float* __restrict__ bt1, const float* __restrict__ m1,
                           const float* __restrict__ v1,
                           const float* __restrict__ b2, const float* __restrict__ g2,
                           const float* __restrict__ bt2, const float* __restrict__ m2,
                           const float* __restrict__ v2,
                           const float* __restrict__ b3, const float* __restrict__ g3,
                           const float* __restrict__ bt3, const float* __restrict__ m3,
                           const float* __restrict__ v3,
                           float* __restrict__ out_pooled) {
    extern __shared__ char smem[];
    const int lane = threadIdx.x & 31;
    const int w    = threadIdx.x >> 5;
    const int cidx = blockIdx.x * 2 + w;
    const int b    = cidx / SS;

    char* base = smem + (size_t)w * 32000;
    _Float16* Fa = (_Float16*)base;        // 64 x 104
    _Float16* H1 = Fa + 64 * 104;          // 64 x 72
    _Float16* H2 = H1 + 64 * 72;           // 64 x 72
    float* rowm  = (float*)(base + 31744); // 64

    const _Float16* wf1 = wf;
    const _Float16* wf2 = wf + 12 * 512;
    const _Float16* wf3 = wf + 20 * 512;

    // keep the 36 KB packed-weight buffer hot: every neighborhood wave rereads it
    for (int off = threadIdx.x * 64; off < 36 * 1024; off += 64 * 64)
        __builtin_prefetch((const char*)wf + off, 0, 1);

    const float* xb  = x  + (size_t)b * NN * 3;
    const float* xcb = xc + (size_t)b * NN * 64;
    const int*   nb  = idx_ball + (size_t)cidx * KK;

    // ---- gather: channels 0..63 = x_complete, 64..66 = rel coords, 67..95 = 0 ----
    int fp = idx_fps[cidx];
    float cenc = (lane < 3) ? xb[fp * 3 + lane] : 0.0f;
    for (int kk = 0; kk < KK; ++kk) {
        int pi = nb[kk];
        int p  = pi < 0 ? 0 : pi;
        const float* f = xcb + (size_t)p * 64;
        Fa[kk * 104 + lane]      = (_Float16)f[lane];
        Fa[kk * 104 + 32 + lane] = (_Float16)f[32 + lane];
        _Float16 tail = (_Float16)0.0f;
        if (lane < 3) tail = (_Float16)(xb[p * 3 + lane] - cenc);
        Fa[kk * 104 + 64 + lane] = tail;   // lanes 3..31 zero-pad cols 67..95
        if (lane == 0) rowm[kk] = (pi < 0) ? 1.0f : 0.0f;
    }
    __syncthreads();

    const int ln = lane & 15;
    const int hs = lane >> 4;

    // ---- layer 1: [64x96(f16)] x [96x64] -> H1, BN+ReLU fused ----
    #pragma unroll 1
    for (int nt = 0; nt < 4; ++nt) {
        v16h bf0 = load_bfrag(wf1, nt * 3 + 0, lane);
        v16h bf1 = load_bfrag(wf1, nt * 3 + 1, lane);
        v16h bf2 = load_bfrag(wf1, nt * 3 + 2, lane);
        int c = nt * 16 + ln;
        float sg = g1[c] * rsqrtf(v1[c] + EPSBN);
        float sh = (b1[c] - m1[c]) * sg + bt1[c];
        #pragma unroll
        for (int mt = 0; mt < 4; ++mt) {
            v16h a0 = load_afrag(Fa, mt * 16 + ln, 104, 0, hs);
            v16h a1 = load_afrag(Fa, mt * 16 + ln, 104, 1, hs);
            v16h a2 = load_afrag(Fa, mt * 16 + ln, 104, 2, hs);
            v8f acc = {};
            acc = wmma32(a0, bf0, acc);
            acc = wmma32(a1, bf1, acc);
            acc = wmma32(a2, bf2, acc);
            #pragma unroll
            for (int q = 0; q < 8; ++q) {
                float o = fmaxf(acc[q] * sg + sh, 0.0f);
                H1[(mt * 16 + hs * 8 + q) * 72 + c] = (_Float16)o;
            }
        }
    }
    __syncthreads();

    // ---- layer 2: [64x64] x [64x64] -> H2 ----
    #pragma unroll 1
    for (int nt = 0; nt < 4; ++nt) {
        v16h bf0 = load_bfrag(wf2, nt * 2 + 0, lane);
        v16h bf1 = load_bfrag(wf2, nt * 2 + 1, lane);
        int c = nt * 16 + ln;
        float sg = g2[c] * rsqrtf(v2[c] + EPSBN);
        float sh = (b2[c] - m2[c]) * sg + bt2[c];
        #pragma unroll
        for (int mt = 0; mt < 4; ++mt) {
            v16h a0 = load_afrag(H1, mt * 16 + ln, 72, 0, hs);
            v16h a1 = load_afrag(H1, mt * 16 + ln, 72, 1, hs);
            v8f acc = {};
            acc = wmma32(a0, bf0, acc);
            acc = wmma32(a1, bf1, acc);
            #pragma unroll
            for (int q = 0; q < 8; ++q) {
                float o = fmaxf(acc[q] * sg + sh, 0.0f);
                H2[(mt * 16 + hs * 8 + q) * 72 + c] = (_Float16)o;
            }
        }
    }
    __syncthreads();

    // ---- layer 3: [64x64] x [64x128] -> masked max over the 64 neighbors ----
    const float NEGINF = -__builtin_inff();

    // compress row-validity (this lane owns rows mt*16 + hs*8 + q) into 32 bits
    unsigned rmbits = 0;
    #pragma unroll 1
    for (int t = 0; t < 32; ++t) {
        int mt = t >> 3, q = t & 7;
        if (rowm[mt * 16 + hs * 8 + q] != 0.0f) rmbits |= (1u << t);
    }

    #pragma unroll 1
    for (int nt = 0; nt < 8; ++nt) {
        v16h bf0 = load_bfrag(wf3, nt * 2 + 0, lane);
        v16h bf1 = load_bfrag(wf3, nt * 2 + 1, lane);
        int c = nt * 16 + ln;
        float sg = g3[c] * rsqrtf(v3[c] + EPSBN);
        float sh = (b3[c] - m3[c]) * sg + bt3[c];
        float cmax = NEGINF;
        #pragma unroll
        for (int mt = 0; mt < 4; ++mt) {
            v16h a0 = load_afrag(H2, mt * 16 + ln, 72, 0, hs);
            v16h a1 = load_afrag(H2, mt * 16 + ln, 72, 1, hs);
            v8f acc = {};
            acc = wmma32(a0, bf0, acc);
            acc = wmma32(a1, bf1, acc);
            #pragma unroll
            for (int q = 0; q < 8; ++q) {
                float o = fmaxf(acc[q] * sg + sh, 0.0f);
                if (rmbits & (1u << (mt * 8 + q))) o = NEGINF;
                cmax = fmaxf(cmax, o);
            }
        }
        float other = __shfl_xor(cmax, 16, 32);  // lanes L and L^16 share the column
        float mx = fmaxf(cmax, other);
        if (lane < 16) out_pooled[(size_t)cidx * 128 + nt * 16 + lane] = mx;
    }
}

// ---------------------------------------------------------------------------
// Host-side launcher. Workspace layout (needs ~2.07 MB):
//   [0, 32768)              idx_fps  : B*S int32
//   [32768, 32768+2097152)  idx_ball : B*S*K int32
//   [+0, +36864)            packed f16 weight fragments (36 tiles * 512 halves)
// Output: [B*S*3 centroids | B*S*128 pooled], f32.
// ---------------------------------------------------------------------------
extern "C" void kernel_launch(void* const* d_in, const int* in_sizes, int n_in,
                              void* d_out, int out_size, void* d_ws, size_t ws_size,
                              hipStream_t stream) {
    (void)in_sizes; (void)n_in; (void)out_size; (void)ws_size;
    const float* x  = (const float*)d_in[0];
    const float* xc = (const float*)d_in[1];
    const float* W1 = (const float*)d_in[2];
    const float* b1 = (const float*)d_in[3];
    const float* g1 = (const float*)d_in[4];
    const float* bt1= (const float*)d_in[5];
    const float* m1 = (const float*)d_in[6];
    const float* v1 = (const float*)d_in[7];
    const float* W2 = (const float*)d_in[8];
    const float* b2 = (const float*)d_in[9];
    const float* g2 = (const float*)d_in[10];
    const float* bt2= (const float*)d_in[11];
    const float* m2 = (const float*)d_in[12];
    const float* v2 = (const float*)d_in[13];
    const float* W3 = (const float*)d_in[14];
    const float* b3 = (const float*)d_in[15];
    const float* g3 = (const float*)d_in[16];
    const float* bt3= (const float*)d_in[17];
    const float* m3 = (const float*)d_in[18];
    const float* v3 = (const float*)d_in[19];

    char* ws = (char*)d_ws;
    int*       idx_fps  = (int*)ws;
    int*       idx_ball = (int*)(ws + (size_t)BB * SS * 4);
    _Float16*  wf       = (_Float16*)(ws + (size_t)BB * SS * 4 + (size_t)BB * SS * KK * 4);

    float* out_cen    = (float*)d_out;
    float* out_pooled = (float*)d_out + (size_t)BB * SS * 3;

    pack_weights_kernel<<<(36 * 512 + 255) / 256, 256, 0, stream>>>(W1, W2, W3, wf);

    size_t fps_lds = (size_t)NN * 4 + 32 * 8 + 16;
    fps_kernel<<<BB, 1024, fps_lds, stream>>>(x, idx_fps);

    ballq_kernel<<<(BB * SS) / 8, 256, 0, stream>>>(x, idx_fps, idx_ball, out_cen);

    mlp_kernel<<<(BB * SS) / 2, 64, 64000, stream>>>(
        x, xc, idx_fps, idx_ball, wf,
        b1, g1, bt1, m1, v1,
        b2, g2, bt2, m2, v2,
        b3, g3, bt3, m3, v3,
        out_pooled);
}
